// MultiTokenPredictor_11536282157173
// MI455X (gfx1250) — matrix operand order repaired and checked by reading, not compiled, not run on hardware
//
#include <hip/hip_runtime.h>
#include <hip/hip_bf16.h>

typedef __attribute__((ext_vector_type(16))) __bf16 v16bf;
typedef __attribute__((ext_vector_type(8)))  float  v8f;
typedef __attribute__((ext_vector_type(4)))  unsigned int u32x4;
typedef __attribute__((ext_vector_type(8)))  int i32x8;
typedef __attribute__((ext_vector_type(4)))  int i32x4;

#define LN_EPS 1e-5f

// Tensor Data Mover path (gfx1250). Guarded so host pass / older toolchains
// fall back to the manual global->LDS pipeline.
#if defined(__AMDGCN__) && __has_builtin(__builtin_amdgcn_tensor_load_to_lds) && \
    __has_builtin(__builtin_amdgcn_s_wait_tensorcnt)
#define ATH_TDM 1
#else
#define ATH_TDM 0
#endif

static __device__ __forceinline__ __bf16 f2bf(float f) {
    unsigned u = __builtin_bit_cast(unsigned, f);
    unsigned r = u + 0x7FFFu + ((u >> 16) & 1u);   // round-to-nearest-even
    unsigned short s = (unsigned short)(r >> 16);
    return __builtin_bit_cast(__bf16, s);
}

#if ATH_TDM
// Issue one TDM descriptor: 2D tile of `rows` x 32 bf16 from global
// (row stride = K elements) into LDS at byte offset lds_off, with hardware
// row padding: +4 DWORDs after every 16 DWORDs => LDS row stride 80B
// (= 40 bf16 elements; 80B = 20 banks, gcd(20,64)=4 -> 16 consecutive rows
// land on 16 distinct banks for the b128 fragment reads).
static __device__ __forceinline__ void tdm_load_tile(const void* gbase,
                                                     unsigned lds_off,
                                                     int rows, int K)
{
    const unsigned long long ga = (unsigned long long)gbase;
    u32x4 g0 = {
        1u,                                   // count=1 (user descriptor)
        lds_off,                              // lds_addr (bytes)
        (unsigned)(ga & 0xFFFFFFFFu),         // global_addr[31:0]
        (unsigned)(ga >> 32) | (2u << 30)     // global_addr[56:32] | type=2
    };
    i32x8 g1 = {
        (int)((1u << 16)                      // data_size = 2 bytes
            | (1u << 20)                      // pad_enable
            | (3u << 22)                      // pad_interval: 16 DWORDs (64B)
            | (3u << 25)),                    // pad_amount: 4 DWORDs (16B)
        (int)(((unsigned)K & 0xFFFFu) << 16), // tensor_dim0[15:0] (abar=0)
        (int)(((unsigned)K >> 16) & 0xFFFFu), // tensor_dim0[31:16] | dim1 lo=0
        (int)(16u | (32u << 16)),             // tensor_dim1 hi (=1M) | tile_dim0=32
        (int)rows,                            // tile_dim1=rows | tile_dim2=0
        (int)K,                               // tensor_dim0_stride[31:0]
        0, 0                                  // stride0 hi, stride1 (unused, 2D)
    };
    i32x4 z4 = {0, 0, 0, 0};
#if defined(__clang_major__) && __clang_major__ >= 23
    i32x8 z8 = {0, 0, 0, 0, 0, 0, 0, 0};
    __builtin_amdgcn_tensor_load_to_lds(g0, g1, z4, z4, z8, 0);
#else
    __builtin_amdgcn_tensor_load_to_lds(g0, g1, z4, z4, 0);
#endif
}

static __device__ __forceinline__ unsigned lds_off_of(const void* p) {
    return (unsigned)(unsigned long long)p;   // generic addr low 32b = LDS offset
}
#endif  // ATH_TDM

// ---------------------------------------------------------------------------
// Row-wise LayerNorm over D=4096 columns, fp32 in -> bf16 out. Block = row.
// ---------------------------------------------------------------------------
__global__ __launch_bounds__(256) void ln_rows_kernel(
    const float* __restrict__ x, const float* __restrict__ gamma,
    const float* __restrict__ beta, __bf16* __restrict__ out, int D)
{
    const int row = blockIdx.x;
    const int t = threadIdx.x;
    const float* xr = x + (size_t)row * D;
    __bf16* orow = out + (size_t)row * D;

    float v[16];
    float s = 0.f;
#pragma unroll
    for (int i = 0; i < 16; ++i) { v[i] = xr[t + i * 256]; s += v[i]; }

    __shared__ float red[256];
    red[t] = s; __syncthreads();
    for (int off = 128; off > 0; off >>= 1) {
        if (t < off) red[t] += red[t + off];
        __syncthreads();
    }
    const float mu = red[0] * (1.f / 4096.f);
    __syncthreads();

    float s2 = 0.f;
#pragma unroll
    for (int i = 0; i < 16; ++i) { float d = v[i] - mu; s2 += d * d; }
    red[t] = s2; __syncthreads();
    for (int off = 128; off > 0; off >>= 1) {
        if (t < off) red[t] += red[t + off];
        __syncthreads();
    }
    const float rstd = rsqrtf(red[0] * (1.f / 4096.f) + LN_EPS);

#pragma unroll
    for (int i = 0; i < 16; ++i) {
        int c = t + i * 256;
        orow[c] = f2bf((v[i] - mu) * rstd * gamma[c] + beta[c]);
    }
}

// ---------------------------------------------------------------------------
// Elementwise fp32 -> bf16 (weights), grid-stride.
// ---------------------------------------------------------------------------
__global__ __launch_bounds__(256) void cvt_f32_bf16_kernel(
    const float* __restrict__ in, __bf16* __restrict__ out, size_t n)
{
    size_t i = (size_t)blockIdx.x * blockDim.x + threadIdx.x;
    const size_t stride = (size_t)gridDim.x * blockDim.x;
    for (; i < n; i += stride) out[i] = f2bf(in[i]);
}

// ---------------------------------------------------------------------------
// Transpose + convert: out[d*D + v] = bf16(in[v*D + d]).  32x32 LDS tiles.
// ---------------------------------------------------------------------------
__global__ __launch_bounds__(256) void transpose_cvt_kernel(
    const float* __restrict__ in, __bf16* __restrict__ out, int D)
{
    __shared__ float tile[32][33];
    const int bx = blockIdx.x * 32, by = blockIdx.y * 32;
    const int tx = threadIdx.x & 31, ty0 = threadIdx.x >> 5;  // 8 rows/pass
#pragma unroll
    for (int ty = ty0; ty < 32; ty += 8)
        tile[ty][tx] = in[(size_t)(by + ty) * D + bx + tx];
    __syncthreads();
#pragma unroll
    for (int ty = ty0; ty < 32; ty += 8)
        out[(size_t)(bx + ty) * D + by + tx] = f2bf(tile[tx][ty]);
}

// ---------------------------------------------------------------------------
// Row-wise softmax over D=4096, fp32 in -> bf16 out. Block = row.
// ---------------------------------------------------------------------------
__global__ __launch_bounds__(256) void softmax_rows_kernel(
    const float* __restrict__ in, __bf16* __restrict__ out, int D)
{
    const int row = blockIdx.x;
    const int t = threadIdx.x;
    const float* xr = in + (size_t)row * D;
    __bf16* orow = out + (size_t)row * D;

    float v[16];
    float m = -3.0e38f;
#pragma unroll
    for (int i = 0; i < 16; ++i) { v[i] = xr[t + i * 256]; m = fmaxf(m, v[i]); }

    __shared__ float red[256];
    red[t] = m; __syncthreads();
    for (int off = 128; off > 0; off >>= 1) {
        if (t < off) red[t] = fmaxf(red[t], red[t + off]);
        __syncthreads();
    }
    m = red[0]; __syncthreads();

    float s = 0.f;
#pragma unroll
    for (int i = 0; i < 16; ++i) { v[i] = __expf(v[i] - m); s += v[i]; }
    red[t] = s; __syncthreads();
    for (int off = 128; off > 0; off >>= 1) {
        if (t < off) red[t] += red[t + off];
        __syncthreads();
    }
    const float inv = 1.f / red[0];

#pragma unroll
    for (int i = 0; i < 16; ++i) orow[t + i * 256] = f2bf(v[i] * inv);
}

// ---------------------------------------------------------------------------
// bf16 WMMA GEMM: C[M,N] = A[M,K] (row-major) x W[N,K]^T (W row-major).
// Workgroup tile 128x256, 8 waves (2x4), wave tile 64x64 (4x4 WMMA accums:
// 16 ds_load_b128 : 16 v_wmma per K-step). K-step 32, double-buffered LDS.
// Global->LDS via TDM (wave0: A, wave1/2: B halves; s_wait_tensorcnt +
// barrier); manual path as fallback.  EPI: 0 = SiLU->bf16, 1 = f32, 2 = bf16.
// ---------------------------------------------------------------------------
union Frag { uint4 q[2]; v16bf v; };

template <int EPI>
__global__ __launch_bounds__(256) void gemm_bf16_kernel(
    const __bf16* __restrict__ A, const __bf16* __restrict__ W,
    void* __restrict__ Cout, int M, int N, int K)
{
    constexpr int LDSS = 40;                       // padded row stride (elems)
    __shared__ __bf16 lA[2][128 * LDSS];           // 20 KB
    __shared__ __bf16 lB[2][256 * LDSS];           // 40 KB

    const int tid  = threadIdx.x;
    const int lane = tid & 31;
    const int wave = tid >> 5;
    const int wm = wave >> 2;                      // 0..1  (M direction)
    const int wn = wave & 3;                       // 0..3  (N direction)
    const int bM = blockIdx.y * 128;
    const int bN = blockIdx.x * 256;

    v8f acc[4][4];
#pragma unroll
    for (int mi = 0; mi < 4; ++mi)
#pragma unroll
        for (int ni = 0; ni < 4; ++ni) acc[mi][ni] = v8f{};

    const int NK = K >> 5;

#if ATH_TDM
    // ---- stage-0 tiles via Tensor Data Mover (3 concurrent 128-row DMAs) ----
    if (wave == 0)
        tdm_load_tile(A + (size_t)bM * K, lds_off_of(&lA[0][0]), 128, K);
    else if (wave == 1)
        tdm_load_tile(W + (size_t)bN * K, lds_off_of(&lB[0][0]), 128, K);
    else if (wave == 2)
        tdm_load_tile(W + (size_t)(bN + 128) * K,
                      lds_off_of(&lB[0][128 * LDSS]), 128, K);
    if (wave < 3) __builtin_amdgcn_s_wait_tensorcnt(0);
    __syncthreads();
#else
    const int lr = tid >> 1;                       // 0..127
    const int lh = tid & 1;                        // which 16-elem half
    const size_t aBase  = (size_t)(bM + lr) * K + lh * 16;
    const size_t bBase0 = (size_t)(bN + lr) * K + lh * 16;
    const size_t bBase1 = (size_t)(bN + lr + 128) * K + lh * 16;
    const int ldsOffA  = lr * LDSS + lh * 16;
    const int ldsOffB1 = (lr + 128) * LDSS + lh * 16;

    uint4 pa0 = *(const uint4*)(A + aBase);
    uint4 pa1 = *(const uint4*)(A + aBase + 8);
    uint4 pb0 = *(const uint4*)(W + bBase0);
    uint4 pb1 = *(const uint4*)(W + bBase0 + 8);
    uint4 pc0 = *(const uint4*)(W + bBase1);
    uint4 pc1 = *(const uint4*)(W + bBase1 + 8);
    *(uint4*)&lA[0][ldsOffA]      = pa0;
    *(uint4*)&lA[0][ldsOffA + 8]  = pa1;
    *(uint4*)&lB[0][ldsOffA]      = pb0;
    *(uint4*)&lB[0][ldsOffA + 8]  = pb1;
    *(uint4*)&lB[0][ldsOffB1]     = pc0;
    *(uint4*)&lB[0][ldsOffB1 + 8] = pc1;
    __syncthreads();
#endif

    const int khalf = lane >> 4;                   // K-half select per ISA map
    const int mrow  = lane & 15;

    for (int kt = 0; kt < NK; ++kt) {
        const int s = kt & 1;
        if (kt + 1 < NK) {                         // prefetch next K-slab
            const size_t ko = (size_t)(kt + 1) * 32;
#if ATH_TDM
            if (wave == 0)
                tdm_load_tile(A + (size_t)bM * K + ko,
                              lds_off_of(&lA[s ^ 1][0]), 128, K);
            else if (wave == 1)
                tdm_load_tile(W + (size_t)bN * K + ko,
                              lds_off_of(&lB[s ^ 1][0]), 128, K);
            else if (wave == 2)
                tdm_load_tile(W + (size_t)(bN + 128) * K + ko,
                              lds_off_of(&lB[s ^ 1][128 * LDSS]), 128, K);
#else
            pa0 = *(const uint4*)(A + aBase + ko);
            pa1 = *(const uint4*)(A + aBase + ko + 8);
            pb0 = *(const uint4*)(W + bBase0 + ko);
            pb1 = *(const uint4*)(W + bBase0 + ko + 8);
            pc0 = *(const uint4*)(W + bBase1 + ko);
            pc1 = *(const uint4*)(W + bBase1 + ko + 8);
#endif
        }

        Frag fa[4], fb[4];
#pragma unroll
        for (int mi = 0; mi < 4; ++mi) {
            const int r = wm * 64 + mi * 16 + mrow;
            // A 16x32 bf16 layout: elems 0..7 -> K = khalf*8..+8,
            //                      elems 8..15 -> K = 16 + khalf*8..+8
            fa[mi].q[0] = *(const uint4*)&lA[s][r * LDSS + khalf * 8];
            fa[mi].q[1] = *(const uint4*)&lA[s][r * LDSS + 16 + khalf * 8];
        }
#pragma unroll
        for (int ni = 0; ni < 4; ++ni) {
            const int c = wn * 64 + ni * 16 + mrow;
            // B 32x16 bf16 layout: elems 0..7 -> K = khalf*8..+8, elems 8..15 -> K = 16 + khalf*8..+8
            fb[ni].q[0] = *(const uint4*)&lB[s][c * LDSS + khalf * 8];
            fb[ni].q[1] = *(const uint4*)&lB[s][c * LDSS + 16 + khalf * 8];
        }

#pragma unroll
        for (int mi = 0; mi < 4; ++mi)
#pragma unroll
            for (int ni = 0; ni < 4; ++ni)
                acc[mi][ni] = __builtin_amdgcn_wmma_f32_16x16x32_bf16(
                    false, fa[mi].v, false, fb[ni].v,
                    (short)0, acc[mi][ni], false, false);

        if (kt + 1 < NK) {
#if ATH_TDM
            if (wave < 3) __builtin_amdgcn_s_wait_tensorcnt(0);
#else
            *(uint4*)&lA[s ^ 1][ldsOffA]      = pa0;
            *(uint4*)&lA[s ^ 1][ldsOffA + 8]  = pa1;
            *(uint4*)&lB[s ^ 1][ldsOffA]      = pb0;
            *(uint4*)&lB[s ^ 1][ldsOffA + 8]  = pb1;
            *(uint4*)&lB[s ^ 1][ldsOffB1]     = pc0;
            *(uint4*)&lB[s ^ 1][ldsOffB1 + 8] = pc1;
#endif
        }
        __syncthreads();
    }

    // epilogue; C/D layout: VGPR r -> row r + (lane>=16)*8, col lane&15
    const int hi = lane >> 4;
#pragma unroll
    for (int mi = 0; mi < 4; ++mi) {
#pragma unroll
        for (int ni = 0; ni < 4; ++ni) {
            const int gn = bN + wn * 64 + ni * 16 + mrow;
#pragma unroll
            for (int r = 0; r < 8; ++r) {
                const int gm = bM + wm * 64 + mi * 16 + hi * 8 + r;
                const float x = acc[mi][ni][r];
                if (EPI == 0) {
#if defined(__AMDGCN__)
                    const float sig = __builtin_amdgcn_rcpf(1.f + __expf(-x));
#else
                    const float sig = 1.f / (1.f + __expf(-x));
#endif
                    ((__bf16*)Cout)[(size_t)gm * N + gn] = f2bf(x * sig);  // SiLU
                } else if (EPI == 1) {
                    ((float*)Cout)[(size_t)gm * N + gn] = x;
                } else {
                    ((__bf16*)Cout)[(size_t)gm * N + gn] = f2bf(x);
                }
            }
        }
    }
}

// ---------------------------------------------------------------------------
extern "C" void kernel_launch(void* const* d_in, const int* in_sizes, int n_in,
                              void* d_out, int out_size, void* d_ws, size_t ws_size,
                              hipStream_t stream)
{
    const float* hs    = (const float*)d_in[0];  // [2,4096,4096]
    const float* gamma = (const float*)d_in[1];  // [4096]
    const float* beta  = (const float*)d_in[2];  // [4096]
    const float* w1    = (const float*)d_in[3];  // [3,4096,4096]
    const float* w2    = (const float*)d_in[4];  // [3,4096,4096]
    float* out = (float*)d_out;                  // 3 x [2,4096,4096] fp32

    const int D = 4096, L = 3;
    const int M = 2 * 4096;                      // B*T = 8192

    char* ws = (char*)d_ws;
    __bf16* hb   = (__bf16*)(ws);                                   // 64 MB
    __bf16* midb = (__bf16*)(ws + (size_t)M * D * 2);               // 64 MB (mid & softmax reuse)
    __bf16* w1b  = (__bf16*)(ws + (size_t)M * D * 2 * 2);           // 96 MB
    __bf16* w2b  = w1b + (size_t)L * D * D;                         // 96 MB
    __bf16* rptb = w2b + (size_t)L * D * D;                         // 32 MB (w1[0]^T)

    // h = LayerNorm(x), to bf16
    ln_rows_kernel<<<M, 256, 0, stream>>>(hs, gamma, beta, hb, D);

    // weights to bf16 (once); reproj = w1[0] transposed so every GEMM is A x W^T
    const size_t wn = (size_t)L * D * D;
    cvt_f32_bf16_kernel<<<2048, 256, 0, stream>>>(w1, w1b, wn);
    cvt_f32_bf16_kernel<<<2048, 256, 0, stream>>>(w2, w2b, wn);
    transpose_cvt_kernel<<<dim3(D / 32, D / 32), 256, 0, stream>>>(w1, rptb, D);

    const dim3 ggrid(D / 256, M / 128);          // (16, 64) = 1024 workgroups
    for (int i = 0; i < L; ++i) {
        // mid = silu(h @ w1[i]^T)
        gemm_bf16_kernel<0><<<ggrid, 256, 0, stream>>>(
            hb, w1b + (size_t)i * D * D, midb, M, D, D);
        // pred = mid @ w2[i]^T  (fp32, straight to output slice)
        gemm_bf16_kernel<1><<<ggrid, 256, 0, stream>>>(
            midb, w2b + (size_t)i * D * D, out + (size_t)i * M * D, M, D, D);
        if (i + 1 < L) {
            // h = softmax(pred) @ w1[0]   (reproj; skipped after last layer)
            softmax_rows_kernel<<<M, 256, 0, stream>>>(
                out + (size_t)i * M * D, midb, D);
            gemm_bf16_kernel<2><<<ggrid, 256, 0, stream>>>(
                midb, rptb, hb, M, D, D);
        }
    }
}
